// IDPSNet_58016418234864
// MI455X (gfx1250) — compile-verified, matrix operand order checked by reading, not run on hardware
//
#include <hip/hip_runtime.h>
#include <hip/hip_bf16.h>

// ---------------------------------------------------------------------------
// IDPSNet for MI455X (gfx1250, wave32, WMMA).
// Bandwidth-bound: ~280 MB HBM traffic => ~12us floor @ 23.3 TB/s.
// GEMM pipeline: double-buffered LDS tiles filled with
// global_load_async_to_lds_b128 (ASYNCcnt), consumed as bf16 fragments into
// v_wmma_f32_16x16x32_bf16 (2 acc/wave), split-K + f32 atomics for the
// K=49152 GEMM so enough waves are live to saturate HBM.
// ---------------------------------------------------------------------------

typedef __attribute__((ext_vector_type(16))) __bf16 v16bf;
typedef __attribute__((ext_vector_type(8)))  float  v8f;

#define B_SZ    8
#define N_PAT   256
#define M_SEL   64
#define K_SEL   16
#define D_DIM   512
#define K_LOW   3072
#define K_HIGH  49152
#define S_NOISE 100
#define N_HEADS 8
#define DH      64

#define MBLK    32
#define NBLK    128

// 16-bit A-matrix 16x32 layout (ISA 7.12.2): lane L holds row m=L%16;
// VGPR v (2 bf16) holds K = {v*2, v*2+1} (+8 for lanes 16-31, +16 for v>=4).
__device__ __forceinline__ int kbase_of(int v, int hi) {
    return (v < 4) ? (v * 2 + hi * 8) : (16 + (v - 4) * 2 + hi * 8);
}

// Async DMA: 16 bytes global -> LDS, per lane (GLOBAL_LOAD_ASYNC_TO_LDS_B128,
// GV addressing; tracked by ASYNCcnt).  vdst = LDS byte offset VGPR.
__device__ __forceinline__ void async_ld_b128(void* lds_dst, const void* gsrc) {
    const unsigned l = (unsigned)(size_t)lds_dst;   // addr[31:0] = LDS offset
    asm volatile("global_load_async_to_lds_b128 %0, %1, off"
                 :: "v"(l), "v"(gsrc) : "memory");
}
__device__ __forceinline__ void wait_async0() {
    asm volatile("s_wait_asynccnt 0" ::: "memory");
}

// ------------------------- generic WMMA GEMM -------------------------------
// C[M,N] = A[M,K] * B[K,N] (+ bias[N]).  Row-major f32 in memory.
// Block = 256 threads (8 waves), computes a 32x128 tile of C.
// SPLITK: each blockIdx.y handles a K-chunk and atomically accumulates into C
// (C must be pre-initialized with the bias).
template <bool SPLITK>
__global__ void gemm_wmma_kernel(const float* __restrict__ A,
                                 const float* __restrict__ Bm,
                                 const float* __restrict__ bias,
                                 float* __restrict__ C,
                                 int M, int N, int K, int kchunk) {
    __shared__ float As[2][MBLK][32];      // 2 x 4 KB
    __shared__ float Bs[2][32][NBLK];      // 2 x 16 KB
    const int nbn  = N / NBLK;
    const int row0 = (blockIdx.x / nbn) * MBLK;
    const int colB = (blockIdx.x % nbn) * NBLK;
    int kbeg = 0, kend = K;
    if (SPLITK) {
        kbeg = blockIdx.y * kchunk;
        kend = kbeg + kchunk; if (kend > K) kend = K;
    }
    const int tid  = threadIdx.x;
    const int wave = tid >> 5;
    const int lane = tid & 31;
    const int m    = lane & 15;          // row within 16-row half-tile
    const int n    = lane & 15;          // col within 16-col wave slice
    const int hi   = lane >> 4;
    const int cw   = wave * 16 + n;      // col within block tile

    // each thread's staging slots (one A float4, four B float4 per tile)
    const int ra = tid >> 3, ca = (tid & 7) * 4;

    auto stage = [&](int buf, int k0) {
        async_ld_b128(&As[buf][ra][ca], A + (size_t)(row0 + ra) * K + k0 + ca);
#pragma unroll
        for (int i = 0; i < 4; ++i) {
            const int idx = tid + i * 256;          // float4 index in tile
            const int r = idx >> 5, c = (idx & 31) * 4;
            async_ld_b128(&Bs[buf][r][c], Bm + (size_t)(k0 + r) * N + colB + c);
        }
    };

    int buf = 0;
    stage(0, kbeg);
    wait_async0();
    __syncthreads();

    v8f acc0 = {}, acc1 = {};
    for (int k0 = kbeg; k0 < kend; k0 += 32) {
        if (k0 + 32 < kend) stage(buf ^ 1, k0 + 32);       // DMA next tile
        if (k0 + 64 < kend)                                // warm L2 further out
            __builtin_prefetch(Bm + (size_t)(k0 + 64 + ra) * N + colB + ca * 4, 0, 1);

        // ---- build bf16 fragments from LDS, 2 WMMA per K-step ----
        v16bf a0, a1, bb;
#pragma unroll
        for (int v = 0; v < 8; ++v) {
            const int kb = kbase_of(v, hi);
            const float2 fa0 = *(const float2*)&As[buf][m][kb];
            const float2 fa1 = *(const float2*)&As[buf][16 + m][kb];
            a0[2 * v] = (__bf16)fa0.x;  a0[2 * v + 1] = (__bf16)fa0.y;
            a1[2 * v] = (__bf16)fa1.x;  a1[2 * v + 1] = (__bf16)fa1.y;
            bb[2 * v]     = (__bf16)Bs[buf][kb][cw];
            bb[2 * v + 1] = (__bf16)Bs[buf][kb + 1][cw];
        }
        acc0 = __builtin_amdgcn_wmma_f32_16x16x32_bf16(false, a0, false, bb,
                                                       (short)0, acc0, false, false);
        acc1 = __builtin_amdgcn_wmma_f32_16x16x32_bf16(false, a1, false, bb,
                                                       (short)0, acc1, false, false);

        wait_async0();          // next tile landed in LDS
        __syncthreads();        // all waves done reading current tile
        buf ^= 1;
    }

    const int col = colB + cw;
    if (SPLITK) {
#pragma unroll
        for (int v = 0; v < 8; ++v) {
            atomicAdd(&C[(size_t)(row0 + v + 8 * hi) * N + col], acc0[v]);
            atomicAdd(&C[(size_t)(row0 + 16 + v + 8 * hi) * N + col], acc1[v]);
        }
    } else {
        const float bb2 = bias ? bias[col] : 0.0f;
#pragma unroll
        for (int v = 0; v < 8; ++v) {
            C[(size_t)(row0 + v + 8 * hi) * N + col]      = acc0[v] + bb2;
            C[(size_t)(row0 + 16 + v + 8 * hi) * N + col] = acc1[v] + bb2;
        }
    }
}

__global__ void init_bias_kernel(float* __restrict__ C,
                                 const float* __restrict__ bias, int M, int N) {
    const int i = blockIdx.x * blockDim.x + threadIdx.x;
    if (i < M * N) C[i] = bias ? bias[i % N] : 0.0f;
}

// --------------- attention scores: attn[row] = E[row,:] . w ---------------
__global__ void rowdot_kernel(const float* __restrict__ E,
                              const float* __restrict__ w,
                              float* __restrict__ out) {
    const int row  = blockIdx.x;
    const int lane = threadIdx.x;
    float sum = 0.0f;
    for (int d = lane; d < D_DIM; d += 32)
        sum += E[(size_t)row * D_DIM + d] * w[d];
    for (int off = 16; off > 0; off >>= 1)
        sum += __shfl_down(sum, off, 32);
    if (lane == 0) out[row] = sum;
}

// --------------- iterative top-64 memory selection (per batch) -------------
__global__ void select_kernel(const float* __restrict__ attn_all,
                              int* __restrict__ sel_idx) {
    const int b = blockIdx.x;
    const int t = threadIdx.x;                    // 128 threads
    __shared__ int   mem[M_SEL];
    __shared__ float ms[M_SEL];
    __shared__ float nsc[128];
    if (t < M_SEL) { mem[t] = t; ms[t] = attn_all[b * N_PAT + t]; }
    __syncthreads();
    for (int it = 0; it < 3; ++it) {
        const int s = M_SEL + it * 64;
        int idx; float sc;
        if (t < M_SEL) { idx = mem[t]; sc = ms[t]; }
        else           { idx = s + (t - M_SEL); sc = attn_all[b * N_PAT + idx]; }
        nsc[t] = sc;
        __syncthreads();
        int rank = 0;                              // top_k order: desc, ties -> lower pos
        for (int j = 0; j < 128; ++j) {
            const float sj = nsc[j];
            rank += (sj > sc || (sj == sc && j < t)) ? 1 : 0;
        }
        __syncthreads();
        if (rank < M_SEL) { mem[rank] = idx; ms[rank] = sc; }
        __syncthreads();
    }
    if (t < M_SEL) sel_idx[b * M_SEL + t] = mem[t];
}

// ------------- min-max normalized scores of selected patches ---------------
__global__ void norm_scores_kernel(const float* __restrict__ attn_all,
                                   const int* __restrict__ sel_idx,
                                   float* __restrict__ nscore) {
    const int b = blockIdx.x;
    const int t = threadIdx.x;                    // 64 threads
    __shared__ float red[M_SEL];
    const float s = attn_all[b * N_PAT + sel_idx[b * M_SEL + t]];
    red[t] = s; __syncthreads();
    for (int off = 32; off > 0; off >>= 1) {
        if (t < off) red[t] = fminf(red[t], red[t + off]);
        __syncthreads();
    }
    const float mn = red[0]; __syncthreads();
    red[t] = s; __syncthreads();
    for (int off = 32; off > 0; off >>= 1) {
        if (t < off) red[t] = fmaxf(red[t], red[t + off]);
        __syncthreads();
    }
    const float mx = red[0];
    nscore[b * M_SEL + t] = (s - mn) / (mx - mn + 1e-5f);
}

// ----------------- perturbed top-k indicators (mean one-hot) ---------------
__global__ void indicators_kernel(const float* __restrict__ nscore,
                                  const float* __restrict__ noise,
                                  float* __restrict__ ind) {
    const int b = blockIdx.x, s = blockIdx.y;
    const int m = threadIdx.x;                    // 64 threads
    __shared__ float p[M_SEL];
    __shared__ int   fl[M_SEL];
    const float val = nscore[b * M_SEL + m]
                    + 0.05f * noise[((size_t)(b * S_NOISE + s)) * M_SEL + m];
    p[m] = val; __syncthreads();
    int rank = 0;
    for (int j = 0; j < M_SEL; ++j) {
        const float pj = p[j];
        rank += (pj > val || (pj == val && j < m)) ? 1 : 0;
    }
    fl[m] = (rank < K_SEL) ? 1 : 0;
    __syncthreads();
    if (fl[m]) {
        int pos = 0;                              // sorted-ascending one-hot row
        for (int j = 0; j < m; ++j) pos += fl[j];
        atomicAdd(&ind[((size_t)(b * K_SEL + pos)) * M_SEL + m], 1.0f / S_NOISE);
    }
}

// ------- sel_high[b,k,:] = sum_m ind[b,k,m] * patches_high[b,sel[m],:] -----
// WMMA gather-GEMM: M=16, K=64, N=49152 per batch. One 16-col tile per wave;
// B loads are wave-coalesced along columns (16 lanes x 4B contiguous).
__global__ void selhigh_wmma_kernel(const float* __restrict__ ind,
                                    const float* __restrict__ PH,
                                    const int* __restrict__ sel_idx,
                                    float* __restrict__ SH) {
    const int b    = blockIdx.y;
    const int wave = threadIdx.x >> 5;
    const int lane = threadIdx.x & 31;
    __shared__ int sel[M_SEL];
    if (threadIdx.x < M_SEL) sel[threadIdx.x] = sel_idx[b * M_SEL + threadIdx.x];
    __syncthreads();
    const int col0 = (blockIdx.x * 8 + wave) << 4;
    const int n  = lane & 15;
    const int m  = lane & 15;
    const int hi = lane >> 4;
    const float* arow = ind + (size_t)b * K_SEL * M_SEL + m * M_SEL;
    v8f acc = {};
    for (int k0 = 0; k0 < M_SEL; k0 += 32) {
        v16bf a, bf;
#pragma unroll
        for (int v = 0; v < 8; ++v) {
            const int kb = kbase_of(v, hi);
            const int kk = k0 + kb;
            a[2 * v]     = (__bf16)arow[kk];
            a[2 * v + 1] = (__bf16)arow[kk + 1];
            const float* r0 = PH + ((size_t)(b * N_PAT + sel[kk]))     * K_HIGH;
            const float* r1 = PH + ((size_t)(b * N_PAT + sel[kk + 1])) * K_HIGH;
            bf[2 * v]     = (__bf16)r0[col0 + n];
            bf[2 * v + 1] = (__bf16)r1[col0 + n];
        }
        acc = __builtin_amdgcn_wmma_f32_16x16x32_bf16(false, a, false, bf,
                                                      (short)0, acc, false, false);
    }
#pragma unroll
    for (int v = 0; v < 8; ++v)
        SH[((size_t)(b * K_SEL + v + 8 * hi)) * K_HIGH + col0 + n] = acc[v];
}

// -------------------- per-(batch,head) attention ---------------------------
__global__ void attention_kernel(const float* __restrict__ q,
                                 const float* __restrict__ k,
                                 const float* __restrict__ v,
                                 float* __restrict__ o) {
    const int b = blockIdx.x, h = blockIdx.y;
    const int t = threadIdx.x;                    // 256 threads
    __shared__ float qs[K_SEL][DH], ks[K_SEL][DH], vs[K_SEL][DH], sm[K_SEL][K_SEL];
    for (int i = t; i < K_SEL * DH; i += 256) {
        const int r = i >> 6, d = i & 63;
        const size_t base = ((size_t)(b * K_SEL + r)) * D_DIM + h * DH + d;
        qs[r][d] = q[base]; ks[r][d] = k[base]; vs[r][d] = v[base];
    }
    __syncthreads();
    {
        const int i = t >> 4, j = t & 15;
        float dot = 0.0f;
        for (int d = 0; d < DH; ++d) dot += qs[i][d] * ks[j][d];
        sm[i][j] = dot * 0.125f;                  // 1/sqrt(64)
    }
    __syncthreads();
    if (t < K_SEL) {
        float mx = sm[t][0];
        for (int j = 1; j < K_SEL; ++j) mx = fmaxf(mx, sm[t][j]);
        float sum = 0.0f;
        for (int j = 0; j < K_SEL; ++j) { const float e = __expf(sm[t][j] - mx); sm[t][j] = e; sum += e; }
        const float inv = 1.0f / sum;
        for (int j = 0; j < K_SEL; ++j) sm[t][j] *= inv;
    }
    __syncthreads();
    for (int i = t; i < K_SEL * DH; i += 256) {
        const int r = i >> 6, d = i & 63;
        float sum = 0.0f;
        for (int j = 0; j < K_SEL; ++j) sum += sm[r][j] * vs[j][d];
        o[((size_t)(b * K_SEL + r)) * D_DIM + h * DH + d] = sum;
    }
}

// --------------- mean over k, classifier head ------------------------------
__global__ void final_kernel(const float* __restrict__ embs_high,
                             const float* __restrict__ oproj,
                             const float* __restrict__ W_cls,
                             const float* __restrict__ b_cls,
                             float* __restrict__ out) {
    const int b = blockIdx.x;
    __shared__ float agg[D_DIM];
    for (int d = threadIdx.x; d < D_DIM; d += blockDim.x) {
        float sum = 0.0f;
        for (int kk = 0; kk < K_SEL; ++kk) {
            const size_t r = ((size_t)(b * K_SEL + kk)) * D_DIM + d;
            sum += embs_high[r] + oproj[r];
        }
        agg[d] = sum * (1.0f / K_SEL);
    }
    __syncthreads();
    if (threadIdx.x < 10) {
        float sum = b_cls[threadIdx.x];
        for (int d = 0; d < D_DIM; ++d) sum += agg[d] * W_cls[d * 10 + threadIdx.x];
        out[b * 10 + threadIdx.x] = sum;
    }
}

// ---------------------------------------------------------------------------
static inline void launch_gemm(const float* A, const float* Bm, const float* bias,
                               float* C, int M, int N, int K, hipStream_t stream) {
    dim3 grid((M / MBLK) * (N / NBLK));
    gemm_wmma_kernel<false><<<grid, 256, 0, stream>>>(A, Bm, bias, C, M, N, K, K);
}

extern "C" void kernel_launch(void* const* d_in, const int* in_sizes, int n_in,
                              void* d_out, int out_size, void* d_ws, size_t ws_size,
                              hipStream_t stream) {
    const float* patches_low  = (const float*)d_in[0];
    const float* patches_high = (const float*)d_in[1];
    const float* noise        = (const float*)d_in[2];
    const float* W_low        = (const float*)d_in[3];
    const float* b_low        = (const float*)d_in[4];
    const float* w_score      = (const float*)d_in[5];
    const float* W_high       = (const float*)d_in[6];
    const float* b_high       = (const float*)d_in[7];
    const float* Wq           = (const float*)d_in[8];
    const float* Wk           = (const float*)d_in[9];
    const float* Wv           = (const float*)d_in[10];
    const float* Wo           = (const float*)d_in[11];
    const float* W_cls        = (const float*)d_in[12];
    const float* b_cls        = (const float*)d_in[13];
    float* out = (float*)d_out;

    // ---- workspace carve (~31 MB) ----
    char* w = (char*)d_ws;
    auto carve = [&](size_t bytes) { char* p = w; w += (bytes + 255) & ~(size_t)255; return p; };
    float* emb_all   = (float*)carve((size_t)B_SZ * N_PAT * D_DIM * 4);   // 4 MB
    float* attn_all  = (float*)carve((size_t)B_SZ * N_PAT * 4);
    int*   sel_idx   = (int*)  carve((size_t)B_SZ * M_SEL * 4);
    float* nscore    = (float*)carve((size_t)B_SZ * M_SEL * 4);
    float* ind       = (float*)carve((size_t)B_SZ * K_SEL * M_SEL * 4);
    float* sel_high  = (float*)carve((size_t)B_SZ * K_SEL * K_HIGH * 4);  // 25 MB
    float* embs_high = (float*)carve((size_t)B_SZ * K_SEL * D_DIM * 4);
    float* qb        = (float*)carve((size_t)B_SZ * K_SEL * D_DIM * 4);
    float* kb        = (float*)carve((size_t)B_SZ * K_SEL * D_DIM * 4);
    float* vb        = (float*)carve((size_t)B_SZ * K_SEL * D_DIM * 4);
    float* ob        = (float*)carve((size_t)B_SZ * K_SEL * D_DIM * 4);
    float* oproj     = (float*)carve((size_t)B_SZ * K_SEL * D_DIM * 4);

    // 1) Embed ALL low-res patches once: (2048 x 3072) @ (3072 x 512) + b_low.
    //    Covers mem_emb, iteration chunks AND embs_low (same patch data).
    launch_gemm(patches_low, W_low, b_low, emb_all,
                B_SZ * N_PAT, D_DIM, K_LOW, stream);

    // 2) attn[row] = emb . w_score (reused by the loop and by normalization).
    rowdot_kernel<<<B_SZ * N_PAT, 32, 0, stream>>>(emb_all, w_score, attn_all);

    // 3) Iterative top-64 selection.
    select_kernel<<<B_SZ, 128, 0, stream>>>(attn_all, sel_idx);

    // 4) Min-max normalized scores of the selected 64.
    norm_scores_kernel<<<B_SZ, M_SEL, 0, stream>>>(attn_all, sel_idx, nscore);

    // 5) Perturbed top-k indicators (mean of sorted one-hots over 100 samples).
    {
        const int n = B_SZ * K_SEL * M_SEL;
        init_bias_kernel<<<(n + 255) / 256, 256, 0, stream>>>(ind, nullptr, B_SZ * K_SEL, M_SEL);
        indicators_kernel<<<dim3(B_SZ, S_NOISE), M_SEL, 0, stream>>>(nscore, noise, ind);
    }

    // 6) sel_high = ind @ gathered patches_high  (16x64 @ 64x49152 per batch).
    selhigh_wmma_kernel<<<dim3(K_HIGH / 16 / 8, B_SZ), 256, 0, stream>>>(
        ind, patches_high, sel_idx, sel_high);

    // 7) embs_high = sel_high @ W_high + b_high   (128 x 49152 x 512).
    //    Split-K x32 so ~4K waves stream the ~126 MB of operands; partials are
    //    combined with global f32 atomics into the bias-initialized C.
    init_bias_kernel<<<(B_SZ * K_SEL * D_DIM + 255) / 256, 256, 0, stream>>>(
        embs_high, b_high, B_SZ * K_SEL, D_DIM);
    {
        dim3 grid((B_SZ * K_SEL / MBLK) * (D_DIM / NBLK), 32);
        gemm_wmma_kernel<true><<<grid, 256, 0, stream>>>(
            sel_high, W_high, nullptr, embs_high,
            B_SZ * K_SEL, D_DIM, K_HIGH, K_HIGH / 32);
    }

    // 8) q,k,v projections (128 x 512 x 512 each).
    launch_gemm(embs_high, Wq, nullptr, qb, B_SZ * K_SEL, D_DIM, D_DIM, stream);
    launch_gemm(embs_high, Wk, nullptr, kb, B_SZ * K_SEL, D_DIM, D_DIM, stream);
    launch_gemm(embs_high, Wv, nullptr, vb, B_SZ * K_SEL, D_DIM, D_DIM, stream);

    // 9) Per-(batch,head) softmax attention.
    attention_kernel<<<dim3(B_SZ, N_HEADS), 256, 0, stream>>>(qb, kb, vb, ob);

    // 10) Output projection.
    launch_gemm(ob, Wo, nullptr, oproj, B_SZ * K_SEL, D_DIM, D_DIM, stream);

    // 11) Residual mean + classifier head.
    final_kernel<<<B_SZ, 256, 0, stream>>>(embs_high, oproj, W_cls, b_cls, out);
}
